// TransformerLayer_76759655514872
// MI455X (gfx1250) — compile-verified, matrix operand order checked by reading, not run on hardware
//
#include <hip/hip_runtime.h>
#include <hip/hip_bf16.h>

// ---------------------------------------------------------------------------
// Fused sampled-attention transformer layer for MI455X (gfx1250, wave32).
// All GEMMs routed through v_wmma_f32_16x16x32_bf16.
// bf16 LDS rows padded by 16B so the 16 lanes of an A-tile load hit 16
// different bank groups (row stride 272B/1040B, i.e. 16B shift per row over
// the 256B bank span) -> conflict-free ds_load_b128.
// ---------------------------------------------------------------------------

typedef __bf16 v16bf __attribute__((ext_vector_type(16)));
typedef __bf16 v8bf  __attribute__((ext_vector_type(8)));
typedef __bf16 v4bf  __attribute__((ext_vector_type(4)));
typedef float  v8f   __attribute__((ext_vector_type(8)));

#define D     128
#define DP    (D + 8)      // padded bf16 row (272 B)
#define H1P   (4*D + 8)    // padded FFN-hidden bf16 row (1040 B)
#define KNBR  16
#define NH    8
#define TILE  16
#define BLOCK 512
#define EPSLN 1e-5f

// packed bf16 weight offsets (in elements) inside d_ws
#define OFF_WQ 0
#define OFF_WK 16384
#define OFF_WV 32768
#define OFF_WO 49152
#define OFF_W1 65536
#define OFF_W2 131072
#define PK_TOTAL 196608   // bf16 elements -> 384 KiB, resident in L2

// ---------------------------------------------------------------------------
// Weight packer: converts fp32 weights (used as x @ W.T) into bf16 B-tiles in
// the per-lane WMMA layout: tile(c,j) = 32x16 (KxN) slab, stored as
// lane-major 32 lanes x 16 bf16 (32 contiguous bytes per lane).
// Lane slot mapping (mirrors the documented 16-bit A-matrix layout):
//   n = lane%16, half = lane/16
//   slot i in [0,7]  -> k = c*32 + half*8 + i
//   slot i in [8,15] -> k = c*32 + 16 + half*8 + (i-8)
// ---------------------------------------------------------------------------
__global__ void pack_weights_kernel(const float* __restrict__ in_proj_w,
                                    const float* __restrict__ out_w,
                                    const float* __restrict__ w1,
                                    const float* __restrict__ w2,
                                    __bf16* __restrict__ pk)
{
    int gid = blockIdx.x * blockDim.x + threadIdx.x;
    if (gid >= PK_TOTAL) return;

    const float* W; int NIN, NOUT, base;
    if      (gid < OFF_WK) { W = in_proj_w;         NIN = D;   NOUT = D;   base = OFF_WQ; }
    else if (gid < OFF_WV) { W = in_proj_w + D*D;   NIN = D;   NOUT = D;   base = OFF_WK; }
    else if (gid < OFF_WO) { W = in_proj_w + 2*D*D; NIN = D;   NOUT = D;   base = OFF_WV; }
    else if (gid < OFF_W1) { W = out_w;             NIN = D;   NOUT = D;   base = OFF_WO; }
    else if (gid < OFF_W2) { W = w1;                NIN = D;   NOUT = 4*D; base = OFF_W1; }
    else                   { W = w2;                NIN = 4*D; NOUT = D;   base = OFF_W2; }

    int p      = gid - base;
    int tile   = p >> 9;          // 512 elements per tile
    int within = p & 511;
    int lane   = within >> 4;
    int slot   = within & 15;
    int numJ   = NOUT >> 4;
    int c      = tile / numJ;
    int j      = tile % numJ;
    int half   = lane >> 4;
    int n      = lane & 15;
    int kl     = (slot < 8) ? (half*8 + slot) : (16 + half*8 + (slot - 8));
    int k      = c*32 + kl;
    // B[k][n] = W[j*16+n][k]   (since y = x @ W.T)
    pk[gid] = (__bf16)W[(size_t)(j*16 + n)*NIN + k];
}

// A-matrix 16x32 bf16 tile loader (per-lane): two contiguous 16B groups.
__device__ inline v16bf load_a(const __bf16* rowbase, int kbase, int half)
{
    v8bf lo = *(const v8bf*)(rowbase + kbase + half*8);
    v8bf hi = *(const v8bf*)(rowbase + kbase + 16 + half*8);
    v16bf a;
#pragma unroll
    for (int i = 0; i < 8; ++i) { a[i] = lo[i]; a[i+8] = hi[i]; }
    return a;
}

__device__ inline float half_sum16(float v)   // sum over a 16-lane half
{
    v += __shfl_xor(v, 1); v += __shfl_xor(v, 2);
    v += __shfl_xor(v, 4); v += __shfl_xor(v, 8);
    return v;
}
__device__ inline float wave_sum32(float v)
{
    v = half_sum16(v); v += __shfl_xor(v, 16); return v;
}

#define WMMA_BF16(A, B, C) \
    __builtin_amdgcn_wmma_f32_16x16x32_bf16(false, (A), false, (B), (short)0, (C), false, false)

// ---------------------------------------------------------------------------
// One 512-thread workgroup (16 wave32) handles a tile of 16 nodes.
// Static LDS ~145 KB -> 2 workgroups resident per WGP (290 KB < 320 KB).
// ---------------------------------------------------------------------------
__global__ __launch_bounds__(BLOCK)
void fused_layer_kernel(const float* __restrict__ x,
                        const int*   __restrict__ samples,
                        const float* __restrict__ in_proj_b,
                        const float* __restrict__ out_b,
                        const float* __restrict__ b1,
                        const float* __restrict__ b2,
                        const float* __restrict__ g1, const float* __restrict__ be1,
                        const float* __restrict__ g2, const float* __restrict__ be2,
                        const __bf16* __restrict__ pk,
                        float* __restrict__ out)
{
    __shared__ __bf16 Xbf[TILE][DP];        //  4.25 KB node rows, bf16 (padded)
    __shared__ float  Xf [TILE][D];         //  8 KB  node rows, fp32 (residual)
    __shared__ __bf16 Nb [TILE][KNBR][DP];  // 68 KB  gathered neighbors (padded)
    __shared__ float  Qf [TILE][D];         //  8 KB  queries       (reused: Z2f)
    __shared__ float  Sc [TILE][NH][KNBR];  //  8 KB  scores / attention weights
    __shared__ float  Ctxf[TILE][D];        //  8 KB  attn context  (reused: Y2f)
    __shared__ __bf16 Cbf[TILE][DP];        //  4.25 KB context bf16 (padded)
    __shared__ float  Yf [TILE][D];         //  8 KB  attn-out part 0 (+bias+res)
    __shared__ __bf16 Ybf[TILE][DP];        //  4.25 KB post-LN1 bf16 (padded)
    __shared__ __bf16 H1s[TILE][H1P];       // 16.25 KB FFN hidden bf16 (padded)
    __shared__ float  Zf [TILE][D];         //  8 KB  FFN-down part 0 (+bias+res)

    float (*Y2f)[D] = Ctxf;   // attn-out partial 1: Ctxf dead after bf16 convert
    float (*Z2f)[D] = Qf;     // FFN-down partial 1: Qf dead after score phase

    const int tid   = threadIdx.x;
    const int wave  = tid >> 5;            // 0..15
    const int lane  = tid & 31;
    const int half  = lane >> 4;
    const int n16   = lane & 15;
    const int node0 = blockIdx.x * TILE;

    // ---- Phase A: load X tile (row = wave, 4 floats per lane) ----
    {
        float4 v = *(const float4*)(x + (size_t)(node0 + wave)*D + lane*4);
        *(float4*)(&Xf[wave][lane*4]) = v;
        v4bf c; c[0]=(__bf16)v.x; c[1]=(__bf16)v.y; c[2]=(__bf16)v.z; c[3]=(__bf16)v.w;
        *(v4bf*)(&Xbf[wave][lane*4]) = c;
    }
    __syncthreads();

    // ---- Phase B: every wave gathers its node's 16 neighbor rows (L2 hits);
    //      waves 0..7 additionally compute the Q tile via WMMA. ----
    {
        const int nd = node0 + wave;
        for (int r = 0; r < KNBR; ++r) {
            int nb = samples[nd*KNBR + r];                // uniform per wave
            float4 v = *(const float4*)(x + (size_t)nb*D + lane*4);
            v4bf c; c[0]=(__bf16)v.x; c[1]=(__bf16)v.y; c[2]=(__bf16)v.z; c[3]=(__bf16)v.w;
            *(v4bf*)(&Nb[wave][r][lane*4]) = c;           // single ds_store_b64
        }
    }
    if (wave < 8) {                                       // Q = X @ Wq.T + bq
        v8f acc = {0,0,0,0,0,0,0,0};
        const __bf16* rowb = &Xbf[n16][0];
#pragma unroll
        for (int c = 0; c < 4; ++c) {
            v16bf a = load_a(rowb, c*32, half);
            v16bf b = *(const v16bf*)(pk + OFF_WQ + ((c*8 + wave) << 9) + lane*16);
            acc = WMMA_BF16(a, b, acc);
        }
        float bq = in_proj_b[wave*16 + n16];
#pragma unroll
        for (int r = 0; r < 8; ++r)
            Qf[8*half + r][wave*16 + n16] = acc[r] + bq;  // [node][dim]
    }
    __syncthreads();

    // ---- Phase C1: K projection + scores.
    // Wave w owns head h = w&7 for nodes t in [(w>>3)*8, +8): the 4 Wk B-tiles
    // for head h stay in registers and are reused across 8 nodes, cutting the
    // redundant L2 weight stream 8x vs a per-node reload.
    {
        const int h  = wave & 7;
        const int t0 = (wave >> 3) * 8;
        v16bf bk0 = *(const v16bf*)(pk + OFF_WK + ((0*8 + h) << 9) + lane*16);
        v16bf bk1 = *(const v16bf*)(pk + OFF_WK + ((1*8 + h) << 9) + lane*16);
        v16bf bk2 = *(const v16bf*)(pk + OFF_WK + ((2*8 + h) << 9) + lane*16);
        v16bf bk3 = *(const v16bf*)(pk + OFF_WK + ((3*8 + h) << 9) + lane*16);
        float bk  = in_proj_b[D + h*16 + n16];
        for (int tt = 0; tt < 8; ++tt) {
            int t = t0 + tt;
            const __bf16* rowb = &Nb[t][n16][0];          // A row = neighbor n16
            v8f acc = {0,0,0,0,0,0,0,0};
            acc = WMMA_BF16(load_a(rowb,  0, half), bk0, acc);
            acc = WMMA_BF16(load_a(rowb, 32, half), bk1, acc);
            acc = WMMA_BF16(load_a(rowb, 64, half), bk2, acc);
            acc = WMMA_BF16(load_a(rowb, 96, half), bk3, acc);
            float qv = Qf[t][h*16 + n16];
#pragma unroll
            for (int r = 0; r < 8; ++r) {                 // D elem (m=8*half+r, n=n16)
                float p = qv * (acc[r] + bk);
                p = half_sum16(p);                        // sum over head_dim (n)
                if (n16 == 0) Sc[t][h][8*half + r] = p * 0.25f;  // 1/sqrt(16)
            }
        }
    }
    __syncthreads();

    // ---- Phase C2: softmax over 16 neighbors; one thread per (node,head) ----
    if (tid < TILE*NH) {
        int t = tid >> 3, h = tid & 7;
        float mx = -1e30f;
#pragma unroll
        for (int m = 0; m < 16; ++m) mx = fmaxf(mx, Sc[t][h][m]);
        float e[16]; float sum = 0.f;
#pragma unroll
        for (int m = 0; m < 16; ++m) { e[m] = __expf(Sc[t][h][m] - mx); sum += e[m]; }
        float inv = 1.f / sum;
#pragma unroll
        for (int m = 0; m < 16; ++m) Sc[t][h][m] = e[m] * inv;
    }
    __syncthreads();

    // ---- Phase C3: V projection + attention-weighted context (same split) ----
    {
        const int h  = wave & 7;
        const int t0 = (wave >> 3) * 8;
        v16bf bv0 = *(const v16bf*)(pk + OFF_WV + ((0*8 + h) << 9) + lane*16);
        v16bf bv1 = *(const v16bf*)(pk + OFF_WV + ((1*8 + h) << 9) + lane*16);
        v16bf bv2 = *(const v16bf*)(pk + OFF_WV + ((2*8 + h) << 9) + lane*16);
        v16bf bv3 = *(const v16bf*)(pk + OFF_WV + ((3*8 + h) << 9) + lane*16);
        float bv  = in_proj_b[2*D + h*16 + n16];
        for (int tt = 0; tt < 8; ++tt) {
            int t = t0 + tt;
            const __bf16* rowb = &Nb[t][n16][0];
            v8f acc = {0,0,0,0,0,0,0,0};
            acc = WMMA_BF16(load_a(rowb,  0, half), bv0, acc);
            acc = WMMA_BF16(load_a(rowb, 32, half), bv1, acc);
            acc = WMMA_BF16(load_a(rowb, 64, half), bv2, acc);
            acc = WMMA_BF16(load_a(rowb, 96, half), bv3, acc);
            float part = 0.f;
#pragma unroll
            for (int r = 0; r < 8; ++r)
                part += Sc[t][h][8*half + r] * (acc[r] + bv);
            part += __shfl_xor(part, 16);                 // combine m halves
            if (half == 0) Ctxf[t][h*16 + n16] = part;
        }
    }
    __syncthreads();
    {   // ctx -> bf16 (4 elems/thread); Ctxf free for reuse afterwards
        int base = tid*4, t = base >> 7, col = base & 127;
        const float* s = &Ctxf[t][col];
        v4bf c; c[0]=(__bf16)s[0]; c[1]=(__bf16)s[1]; c[2]=(__bf16)s[2]; c[3]=(__bf16)s[3];
        *(v4bf*)(&Cbf[t][col]) = c;
    }
    __syncthreads();

    // ---- Phase D: attn_out = Ctx @ Wo.T + bo + residual.
    // All 16 waves: group g = wave>>3 takes K-chunks {2g, 2g+1} of chunk j. ----
    {
        const int grp = wave >> 3;
        const int j   = wave & 7;
        v8f acc = {0,0,0,0,0,0,0,0};
        const __bf16* rowb = &Cbf[n16][0];
#pragma unroll
        for (int cc = 0; cc < 2; ++cc) {
            int c = grp*2 + cc;
            v16bf a = load_a(rowb, c*32, half);
            v16bf b = *(const v16bf*)(pk + OFF_WO + ((c*8 + j) << 9) + lane*16);
            acc = WMMA_BF16(a, b, acc);
        }
        int colw = j*16 + n16;
        if (grp == 0) {
            float bo = out_b[colw];
#pragma unroll
            for (int r = 0; r < 8; ++r) {
                int m = 8*half + r;
                Yf[m][colw] = acc[r] + bo + Xf[m][colw];
            }
        } else {
#pragma unroll
            for (int r = 0; r < 8; ++r) Y2f[8*half + r][colw] = acc[r];
        }
    }
    __syncthreads();

    // ---- LN1 (wave t -> node t), combining the two attn-out partials ----
    {
        int col = lane*4; float v0[4], s = 0.f, s2 = 0.f;
#pragma unroll
        for (int i = 0; i < 4; ++i) {
            v0[i] = Yf[wave][col+i] + Y2f[wave][col+i];
            s += v0[i]; s2 += v0[i]*v0[i];
        }
        s = wave_sum32(s); s2 = wave_sum32(s2);
        float mean = s * (1.f/128.f);
        float var  = s2 * (1.f/128.f) - mean*mean;
        float rs   = rsqrtf(var + EPSLN);
        v4bf c;
#pragma unroll
        for (int i = 0; i < 4; ++i) {
            float y = (v0[i] - mean) * rs * g1[col+i] + be1[col+i];
            Yf[wave][col+i] = y;                 // fp32 kept for FFN residual
            c[i] = (__bf16)y;
        }
        *(v4bf*)(&Ybf[wave][col]) = c;
    }
    __syncthreads();

    // ---- FFN up: h = relu(Y @ w1.T + b1); 32 N-chunks, 2 per wave ----
    {
        const __bf16* rowb = &Ybf[n16][0];
        for (int jj = 0; jj < 2; ++jj) {
            int j = wave*2 + jj;
            v8f acc = {0,0,0,0,0,0,0,0};
#pragma unroll
            for (int c = 0; c < 4; ++c) {
                v16bf a = load_a(rowb, c*32, half);
                v16bf b = *(const v16bf*)(pk + OFF_W1 + ((c*32 + j) << 9) + lane*16);
                acc = WMMA_BF16(a, b, acc);
            }
            float bb = b1[j*16 + n16];
#pragma unroll
            for (int r = 0; r < 8; ++r) {
                float hv = fmaxf(acc[r] + bb, 0.f);
                H1s[8*half + r][j*16 + n16] = (__bf16)hv;
            }
        }
    }
    __syncthreads();

    // ---- FFN down: z = H1 @ w2.T + b2 + Y.
    // K=512 split across wave groups: group g does K-chunks [8g, 8g+8). ----
    {
        const int grp = wave >> 3;
        const int j   = wave & 7;
        v8f acc = {0,0,0,0,0,0,0,0};
        const __bf16* rowb = &H1s[n16][0];
#pragma unroll
        for (int cc = 0; cc < 8; ++cc) {
            int c = grp*8 + cc;
            v16bf a = load_a(rowb, c*32, half);
            v16bf b = *(const v16bf*)(pk + OFF_W2 + ((c*8 + j) << 9) + lane*16);
            acc = WMMA_BF16(a, b, acc);
        }
        int colw = j*16 + n16;
        if (grp == 0) {
            float bb = b2[colw];
#pragma unroll
            for (int r = 0; r < 8; ++r) {
                int m = 8*half + r;
                Zf[m][colw] = acc[r] + bb + Yf[m][colw];
            }
        } else {
#pragma unroll
            for (int r = 0; r < 8; ++r) Z2f[8*half + r][colw] = acc[r];
        }
    }
    __syncthreads();

    // ---- LN2 + store (wave t -> node t) ----
    {
        int col = lane*4; float v0[4], s = 0.f, s2 = 0.f;
#pragma unroll
        for (int i = 0; i < 4; ++i) {
            v0[i] = Zf[wave][col+i] + Z2f[wave][col+i];
            s += v0[i]; s2 += v0[i]*v0[i];
        }
        s = wave_sum32(s); s2 = wave_sum32(s2);
        float mean = s * (1.f/128.f);
        float var  = s2 * (1.f/128.f) - mean*mean;
        float rs   = rsqrtf(var + EPSLN);
        float* o = out + (size_t)(node0 + wave)*D + col;
#pragma unroll
        for (int i = 0; i < 4; ++i)
            o[i] = (v0[i] - mean) * rs * g2[col+i] + be2[col+i];
    }
}

// ---------------------------------------------------------------------------
extern "C" void kernel_launch(void* const* d_in, const int* in_sizes, int n_in,
                              void* d_out, int out_size, void* d_ws, size_t ws_size,
                              hipStream_t stream)
{
    const float* x         = (const float*)d_in[0];
    const int*   samp      = (const int*)  d_in[1];
    const float* in_proj_w = (const float*)d_in[2];
    const float* in_proj_b = (const float*)d_in[3];
    const float* out_w     = (const float*)d_in[4];
    const float* out_b     = (const float*)d_in[5];
    const float* w1        = (const float*)d_in[6];
    const float* b1        = (const float*)d_in[7];
    const float* w2        = (const float*)d_in[8];
    const float* b2        = (const float*)d_in[9];
    const float* g1        = (const float*)d_in[10];
    const float* be1       = (const float*)d_in[11];
    const float* g2        = (const float*)d_in[12];
    const float* be2       = (const float*)d_in[13];

    __bf16* pk = (__bf16*)d_ws;   // needs PK_TOTAL*2 = 384 KiB of workspace
    int N = in_sizes[0] / D;

    pack_weights_kernel<<<PK_TOTAL/256, 256, 0, stream>>>(in_proj_w, out_w, w1, w2, pk);

    int ntiles = (N + TILE - 1) / TILE;   // N = 100000 -> 6250 exact tiles
    fused_layer_kernel<<<ntiles, BLOCK, 0, stream>>>(
        x, samp, in_proj_b, out_b, b1, b2, g1, be1, g2, be2, pk, (float*)d_out);
}
